// SelectiveSSMBlock_63153199120498
// MI455X (gfx1250) — matrix operand order repaired and checked by reading, not compiled, not run on hardware
//
#include <hip/hip_runtime.h>
#include <stdint.h>

// Problem constants (from reference)
#define BB      2
#define TT      1024
#define DMODEL  1024
#define DI      2048          // d_inner
#define DS      16            // d_state
#define DCONV   4
#define NCHUNK  8
#define CLEN    128           // TT / NCHUNK
#define NTOK    (BB * TT)     // 2048 rows for all GEMMs

typedef __bf16 bf16;
typedef __attribute__((ext_vector_type(16))) __bf16 v16bf;
typedef __attribute__((ext_vector_type(8)))  float  v8f;

__device__ __forceinline__ uint16_t f2bf(float f) {
  union { float f; uint32_t u; } v; v.f = f;
  uint32_t u = v.u;
  return (uint16_t)((u + 0x7FFFu + ((u >> 16) & 1u)) >> 16);  // RNE
}
__device__ __forceinline__ float sigmoidf_(float x) { return 1.0f / (1.0f + expf(-x)); }
__device__ __forceinline__ float siluf_(float x)    { return x / (1.0f + expf(-x)); }

// 32-byte WMMA operand fragment (16 bf16) held in two b128 registers.
struct Frag { uint4 lo, hi; };
__device__ __forceinline__ Frag ldfrag(const uint16_t* p, int hielems) {
  Frag f;
  f.lo = *(const uint4*)(p);
  f.hi = *(const uint4*)(p + hielems);
  return f;
}
__device__ __forceinline__ v16bf fv(const Frag& f) {
  union { Frag f; v16bf v; } u; u.f = f; return u.v;
}

// ---------------------------------------------------------------------------
// Pack fp32 weight W[K,N] (row-major) into WMMA B-fragment layout:
//   out[((kt*ntiles + nt)*32 + lane)*16 + i]  where h = lane>=16,
//   K = kt*32 + (i<8 ? 8h + i : 16 + 8h + i-8),  N = nt*16 + (lane&15)
// Matches A: lanes 0-15 hold K=0..7,16..23; lanes 16-31 K=8..15,24..31.
// ---------------------------------------------------------------------------
__global__ void pack_b_kernel(const float* __restrict__ W, uint16_t* __restrict__ out,
                              int K, int N) {
  int idx = blockIdx.x * blockDim.x + threadIdx.x;
  if (idx >= K * N) return;
  int i  = idx & 15;
  int l  = (idx >> 4) & 31;
  int t  = idx >> 9;
  int ntiles = N >> 4;
  int nt = t % ntiles;
  int kt = t / ntiles;
  int k  = kt * 32 + ((l < 16) ? 0 : 8) + ((i < 8) ? i : 8 + i);
  int n  = nt * 16 + (l & 15);
  out[idx] = f2bf(W[(size_t)k * N + n]);
}

// ---------------------------------------------------------------------------
// LayerNorm over DMODEL per token, output bf16 row-major [NTOK, DMODEL]
// ---------------------------------------------------------------------------
__global__ void __launch_bounds__(256)
ln_kernel(const float* __restrict__ x, const float* __restrict__ g,
          const float* __restrict__ b, uint16_t* __restrict__ xn) {
  const int row  = blockIdx.x;
  const int lane = threadIdx.x & 31;
  const int wave = threadIdx.x >> 5;
  const float* xr = x + (size_t)row * DMODEL;
  __shared__ float shs[8], shs2[8];

  float s = 0.f, s2 = 0.f;
  for (int i = threadIdx.x; i < DMODEL; i += 256) {
    float v = xr[i]; s += v; s2 += v * v;
  }
  #pragma unroll
  for (int o = 16; o > 0; o >>= 1) { s += __shfl_down(s, o); s2 += __shfl_down(s2, o); }
  if (lane == 0) { shs[wave] = s; shs2[wave] = s2; }
  __syncthreads();
  if (threadIdx.x == 0) {
    float S = 0.f, S2 = 0.f;
    #pragma unroll
    for (int w = 0; w < 8; ++w) { S += shs[w]; S2 += shs2[w]; }
    shs[0] = S; shs2[0] = S2;
  }
  __syncthreads();
  const float mu  = shs[0] * (1.0f / DMODEL);
  const float var = shs2[0] * (1.0f / DMODEL) - mu * mu;
  const float inv = rsqrtf(var + 1e-5f);
  for (int i = threadIdx.x; i < DMODEL; i += 256) {
    float v = (xr[i] - mu) * inv * g[i] + b[i];
    xn[(size_t)row * DMODEL + i] = f2bf(v);
  }
}

// ---------------------------------------------------------------------------
// WMMA bf16 GEMM, NT output tiles (16x16) per wave along N, with explicit
// double-buffered operands so step i+1 loads are independent of step i WMMAs
// (lets the scheduler use partial s_wait_loadcnt instead of full drains):
//   C[M,N] = A[M,K] @ B[K,N] + bias[N] (+ residual[M,N])
// A: bf16 row-major. Bp: pre-packed B fragments. K must be a multiple of 64.
// Grids are exact -> EXEC all-ones for every WMMA (ISA requirement).
// ---------------------------------------------------------------------------
template <int NT>
__global__ void __launch_bounds__(256)
wmma_gemm_bf16(const uint16_t* __restrict__ A, const uint16_t* __restrict__ Bp,
               const float* __restrict__ bias, const float* __restrict__ residual,
               float* __restrict__ C, int M, int N, int K) {
  const int lane    = threadIdx.x & 31;
  const int wave    = threadIdx.x >> 5;
  const int ntiles  = N >> 4;
  const int ngroups = ntiles / NT;                 // groups of NT n-tiles
  const int grp     = blockIdx.x * 8 + wave;
  const int mt      = grp / ngroups;
  const int nt0     = (grp - mt * ngroups) * NT;

  // A fragment addressing (ISA 16-bit A 16x32 layout):
  //   lane<16: row=lane,    K in {kk+0..7,  kk+16..23}
  //   lane>=16: row=lane-16, K in {kk+8..15, kk+24..31}
  const int arow  = (mt << 4) + (lane & 15);
  const int akoff = (lane < 16) ? 0 : 8;
  const uint16_t* aptr = A + (size_t)arow * K + akoff;
  const uint16_t* bptr = Bp + ((size_t)nt0 * 32 + lane) * 16;
  const size_t bstride = (size_t)ntiles * 32 * 16;   // elements per kt step

  v8f acc[NT];
  #pragma unroll
  for (int j = 0; j < NT; ++j) acc[j] = (v8f){};

  // Double buffers
  Frag fa0, fa1, fb0[NT], fb1[NT];

  // Prologue: fill buffer 0 (K-step 0)
  fa0 = ldfrag(aptr, 16);
  #pragma unroll
  for (int j = 0; j < NT; ++j) fb0[j] = ldfrag(bptr + (size_t)j * 32 * 16, 8);

  for (int kk = 0; kk < K; kk += 64) {
    // Load buffer 1 (K-step kk+32) -- independent of buffer-0 WMMAs below
    const uint16_t* a1 = aptr + 32;
    const uint16_t* b1 = bptr + bstride;
    fa1 = ldfrag(a1, 16);
    #pragma unroll
    for (int j = 0; j < NT; ++j) fb1[j] = ldfrag(b1 + (size_t)j * 32 * 16, 8);
    __builtin_prefetch(a1 + 32, 0, 1);               // -> global_prefetch_b8
    __builtin_prefetch(b1 + bstride, 0, 1);

    // Compute with buffer 0
    #pragma unroll
    for (int j = 0; j < NT; ++j)
      acc[j] = __builtin_amdgcn_wmma_f32_16x16x32_bf16(
          false, fv(fa0), false, fv(fb0[j]), (short)0, acc[j], false, false);

    aptr += 64;
    bptr += 2 * bstride;

    // Refill buffer 0 (K-step kk+64) unless this was the last pair
    if (kk + 64 < K) {
      fa0 = ldfrag(aptr, 16);
      #pragma unroll
      for (int j = 0; j < NT; ++j) fb0[j] = ldfrag(bptr + (size_t)j * 32 * 16, 8);
    }

    // Compute with buffer 1
    #pragma unroll
    for (int j = 0; j < NT; ++j)
      acc[j] = __builtin_amdgcn_wmma_f32_16x16x32_bf16(
          false, fv(fa1), false, fv(fb1[j]), (short)0, acc[j], false, false);
  }

  // C/D layout: lane<16 -> N=lane, M=mt*16+r ; lane>=16 -> N=lane-16, M=mt*16+8+r
  const int mbase = (mt << 4) + ((lane < 16) ? 0 : 8);
  #pragma unroll
  for (int j = 0; j < NT; ++j) {
    const int ncol = ((nt0 + j) << 4) + (lane & 15);
    const float bv = bias[ncol];
    #pragma unroll
    for (int r = 0; r < 8; ++r) {
      size_t off = (size_t)(mbase + r) * N + ncol;
      float v = acc[j][r] + bv;
      if (residual) v += residual[off];
      C[off] = v;
    }
  }
}

// ---------------------------------------------------------------------------
// Depthwise causal conv1d (k=4) + bias + SiLU over xp half of proj.
// proj: [NTOK, 2*DI]; writes xc fp32 and bf16, both [NTOK, DI].
// ---------------------------------------------------------------------------
__global__ void conv_silu_kernel(const float* __restrict__ proj,
                                 const float* __restrict__ cw,
                                 const float* __restrict__ cb,
                                 float* __restrict__ xc,
                                 uint16_t* __restrict__ xcb) {
  int idx = blockIdx.x * blockDim.x + threadIdx.x;
  if (idx >= NTOK * DI) return;
  int n  = idx & (DI - 1);
  int bt = idx / DI;
  int t  = bt & (TT - 1);
  int b  = bt / TT;
  float acc = cb[n];
  #pragma unroll
  for (int k = 0; k < DCONV; ++k) {
    int tt = t - (DCONV - 1) + k;
    if (tt >= 0)
      acc += cw[n * DCONV + k] * proj[(size_t)(b * TT + tt) * (2 * DI) + n];
  }
  float s = siluf_(acc);
  xc[idx]  = s;
  xcb[idx] = f2bf(s);
}

// ---------------------------------------------------------------------------
// Scan phase A: per-(b,chunk,channel) local scan from zero state.
// ---------------------------------------------------------------------------
__global__ void scan_local_kernel(const float* __restrict__ xc,
                                  const float* __restrict__ Bt,
                                  const float* __restrict__ Amat,
                                  float* __restrict__ localfin) {
  int idx = blockIdx.x * blockDim.x + threadIdx.x;   // B*NCHUNK*DI
  if (idx >= BB * NCHUNK * DI) return;
  int n = idx & (DI - 1);
  int c = (idx / DI) & (NCHUNK - 1);
  int b = idx / (DI * NCHUNK);

  float a[DS], h[DS];
  #pragma unroll
  for (int s = 0; s < DS; ++s) { a[s] = sigmoidf_(Amat[n * DS + s]); h[s] = 0.f; }

  int t0 = c * CLEN;
  for (int t = t0; t < t0 + CLEN; ++t) {
    size_t r = (size_t)(b * TT + t);
    float xv = xc[r * DI + n];
    #pragma unroll
    for (int s = 0; s < DS; ++s) h[s] = a[s] * h[s] + Bt[r * DS + s] * xv;
  }
  size_t o = ((size_t)(b * NCHUNK + c) * DI + n) * DS;
  #pragma unroll
  for (int s = 0; s < DS; ++s) localfin[o + s] = h[s];
}

// ---------------------------------------------------------------------------
// Scan phase B: sequential carry combine across chunks using closed-form
// chunk decay a^CLEN (a is time-invariant -> 7 squarings for CLEN=128).
// ---------------------------------------------------------------------------
__global__ void scan_carry_kernel(const float* __restrict__ localfin,
                                  const float* __restrict__ Amat,
                                  float* __restrict__ carry) {
  int idx = blockIdx.x * blockDim.x + threadIdx.x;   // B*DI
  if (idx >= BB * DI) return;
  int n = idx & (DI - 1);
  int b = idx / DI;

  float aL[DS], h[DS];
  #pragma unroll
  for (int s = 0; s < DS; ++s) {
    float p = sigmoidf_(Amat[n * DS + s]);
    #pragma unroll
    for (int j = 0; j < 7; ++j) p *= p;              // a^(2^7) = a^128
    aL[s] = p; h[s] = 0.f;
  }
  for (int c = 0; c < NCHUNK; ++c) {
    size_t o = ((size_t)(b * NCHUNK + c) * DI + n) * DS;
    #pragma unroll
    for (int s = 0; s < DS; ++s) carry[o + s] = h[s];
    #pragma unroll
    for (int s = 0; s < DS; ++s) h[s] = aL[s] * h[s] + localfin[o + s];
  }
}

// ---------------------------------------------------------------------------
// Scan phase C: re-run local scan seeded with carry; fuse C-readout,
// D skip-connection and SiLU gating; emit bf16 y for out_proj.
// ---------------------------------------------------------------------------
__global__ void scan_apply_kernel(const float* __restrict__ xc,
                                  const float* __restrict__ Bt,
                                  const float* __restrict__ Ct,
                                  const float* __restrict__ Amat,
                                  const float* __restrict__ Dv,
                                  const float* __restrict__ proj,   // gate half
                                  const float* __restrict__ carry,
                                  uint16_t* __restrict__ yb) {
  int idx = blockIdx.x * blockDim.x + threadIdx.x;   // B*NCHUNK*DI
  if (idx >= BB * NCHUNK * DI) return;
  int n = idx & (DI - 1);
  int c = (idx / DI) & (NCHUNK - 1);
  int b = idx / (DI * NCHUNK);

  float a[DS], h[DS];
  size_t o = ((size_t)(b * NCHUNK + c) * DI + n) * DS;
  #pragma unroll
  for (int s = 0; s < DS; ++s) { a[s] = sigmoidf_(Amat[n * DS + s]); h[s] = carry[o + s]; }
  const float dn = Dv[n];

  int t0 = c * CLEN;
  for (int t = t0; t < t0 + CLEN; ++t) {
    size_t r = (size_t)(b * TT + t);
    float xv = xc[r * DI + n];
    float y = 0.f;
    #pragma unroll
    for (int s = 0; s < DS; ++s) {
      h[s] = a[s] * h[s] + Bt[r * DS + s] * xv;
      y += h[s] * Ct[r * DS + s];
    }
    y += dn * xv;
    float g = proj[r * (2 * DI) + DI + n];
    y *= siluf_(g);
    yb[r * DI + n] = f2bf(y);
  }
}

// ---------------------------------------------------------------------------
extern "C" void kernel_launch(void* const* d_in, const int* in_sizes, int n_in,
                              void* d_out, int out_size, void* d_ws, size_t ws_size,
                              hipStream_t stream) {
  (void)in_sizes; (void)n_in; (void)out_size; (void)ws_size;
  const float* x      = (const float*)d_in[0];
  const float* ln_g   = (const float*)d_in[1];
  const float* ln_b   = (const float*)d_in[2];
  const float* W_in   = (const float*)d_in[3];
  const float* b_in   = (const float*)d_in[4];
  const float* conv_w = (const float*)d_in[5];
  const float* conv_b = (const float*)d_in[6];
  const float* Amat   = (const float*)d_in[7];
  const float* W_B    = (const float*)d_in[8];
  const float* b_B    = (const float*)d_in[9];
  const float* W_C    = (const float*)d_in[10];
  const float* b_C    = (const float*)d_in[11];
  const float* Dv     = (const float*)d_in[12];
  const float* W_out  = (const float*)d_in[13];
  const float* b_out  = (const float*)d_in[14];
  float* out = (float*)d_out;

  char* p = (char*)d_ws;
  auto alloc = [&](size_t bytes) -> char* {
    char* r = p; p += (bytes + 255) & ~(size_t)255; return r;
  };
  uint16_t* xn     = (uint16_t*)alloc((size_t)NTOK * DMODEL * 2);     // 4 MB
  uint16_t* Winp   = (uint16_t*)alloc((size_t)DMODEL * 2 * DI * 2);   // 8 MB
  float*    proj   = (float*)   alloc((size_t)NTOK * 2 * DI * 4);     // 32 MB
  float*    xc     = (float*)   alloc((size_t)NTOK * DI * 4);         // 16 MB
  uint16_t* xcb    = (uint16_t*)alloc((size_t)NTOK * DI * 2);         // 8 MB
  uint16_t* WBp    = (uint16_t*)alloc((size_t)DI * DS * 2);
  uint16_t* WCp    = (uint16_t*)alloc((size_t)DI * DS * 2);
  float*    Btb    = (float*)   alloc((size_t)NTOK * DS * 4);
  float*    Ctb    = (float*)   alloc((size_t)NTOK * DS * 4);
  uint16_t* Woutp  = (uint16_t*)alloc((size_t)DI * DMODEL * 2);       // 4 MB
  uint16_t* yb     = (uint16_t*)alloc((size_t)NTOK * DI * 2);         // 8 MB
  float*    locfin = (float*)   alloc((size_t)BB * NCHUNK * DI * DS * 4); // 2 MB
  float*    carry  = (float*)   alloc((size_t)BB * NCHUNK * DI * DS * 4); // 2 MB

  // 1) Weight packing (fp32 -> bf16 WMMA B-fragment layout)
  pack_b_kernel<<<(DMODEL * 2 * DI + 255) / 256, 256, 0, stream>>>(W_in,  Winp,  DMODEL, 2 * DI);
  pack_b_kernel<<<(DI * DS + 255) / 256,          256, 0, stream>>>(W_B,   WBp,   DI, DS);
  pack_b_kernel<<<(DI * DS + 255) / 256,          256, 0, stream>>>(W_C,   WCp,   DI, DS);
  pack_b_kernel<<<(DI * DMODEL + 255) / 256,      256, 0, stream>>>(W_out, Woutp, DI, DMODEL);

  // 2) LayerNorm -> bf16 activations
  ln_kernel<<<NTOK, 256, 0, stream>>>(x, ln_g, ln_b, xn);

  // 3) in_proj GEMM: [2048,1024] @ [1024,4096]; 16x64 strip per wave
  //    groups = (2048/16) * (4096/16/4) = 128*64 = 8192 -> 1024 blocks
  wmma_gemm_bf16<4><<<(NTOK / 16) * (2 * DI / 16 / 4) / 8, 256, 0, stream>>>(
      xn, Winp, b_in, nullptr, proj, NTOK, 2 * DI, DMODEL);

  // 4) depthwise causal conv + SiLU
  conv_silu_kernel<<<(NTOK * DI + 255) / 256, 256, 0, stream>>>(proj, conv_w, conv_b, xc, xcb);

  // 5) B/C projections: [2048,2048] @ [2048,16] (single n-tile -> NT=1)
  wmma_gemm_bf16<1><<<(NTOK / 16) * (DS / 16) / 8, 256, 0, stream>>>(
      xcb, WBp, b_B, nullptr, Btb, NTOK, DS, DI);
  wmma_gemm_bf16<1><<<(NTOK / 16) * (DS / 16) / 8, 256, 0, stream>>>(
      xcb, WCp, b_C, nullptr, Ctb, NTOK, DS, DI);

  // 6) chunked scan (3 phases; a time-invariant -> closed-form chunk decay)
  scan_local_kernel<<<(BB * NCHUNK * DI + 255) / 256, 256, 0, stream>>>(xc, Btb, Amat, locfin);
  scan_carry_kernel<<<(BB * DI + 255) / 256,          256, 0, stream>>>(locfin, Amat, carry);
  scan_apply_kernel<<<(BB * NCHUNK * DI + 255) / 256, 256, 0, stream>>>(
      xc, Btb, Ctb, Amat, Dv, proj, carry, yb);

  // 7) out_proj GEMM + bias + residual: [2048,2048] @ [2048,1024] (+x)
  //    groups = 128 * (1024/16/4) = 128*16 = 2048 -> 256 blocks
  wmma_gemm_bf16<4><<<(NTOK / 16) * (DMODEL / 16 / 4) / 8, 256, 0, stream>>>(
      yb, Woutp, b_out, x, out, NTOK, DMODEL, DI);
}
